// AttentionRing_7138235646301
// MI455X (gfx1250) — matrix operand order repaired and checked by reading, not compiled, not run on hardware
//
#include <hip/hip_runtime.h>

// ---------------------------------------------------------------------------
// AttentionRing on MI455X (gfx1250, wave32).
//   Kernel 1: score[b,n,k] via bf16 WMMA (v_wmma_f32_16x16x32_bf16)
//   Kernel 2: context[b,n,:] = sum_k (score/sum) * values[gather]
// ---------------------------------------------------------------------------

#define BATCH 4
#define NPTS  40962
#define KNEI  7
#define CDIM  128
#define TOTAL_ROWS (BATCH * NPTS * KNEI)   // 1,146,936 flattened (b,n,k) rows
#define NODES      (BATCH * NPTS)          // 163,848

typedef __bf16 bf16;
typedef __attribute__((ext_vector_type(16))) __bf16 v16bf;
typedef __attribute__((ext_vector_type(8)))  __bf16 v8bf;
typedef __attribute__((ext_vector_type(8)))  float  v8f;
typedef __attribute__((ext_vector_type(4)))  float  v4f;

// LDS row pitch for W1^T (bf16): 136 elems = 272 B = 68 banks -> 4-bank
// rotation per row (no 16-way conflicts of the natural 256 B pitch), and
// 272 = 17*16 keeps every v8bf load 16-byte aligned.
#define LDS_STRIDE 136

__device__ __forceinline__ float fast_tanh(float x) {
#if __has_builtin(__builtin_amdgcn_tanhf)
  return __builtin_amdgcn_tanhf(x);
#else
  float y;
  // CDNA5 lists TANH among transcendental VALU ops; v_nop covers the
  // 1-op TRANS result hazard since the compiler can't see inside the asm.
  asm volatile("v_tanh_f32 %0, %1\n\tv_nop" : "=v"(y) : "v"(x));
  return y;
#endif
}

__device__ __forceinline__ float fast_sigmoid(float x) {
  // sigmoid(x) = 0.5*tanh(0.5x) + 0.5  (reuses the HW transcendental)
  return fmaf(0.5f, fast_tanh(0.5f * x), 0.5f);
}

// ---------------------------------------------------------------------------
// Kernel 1: per flattened row r=(b,n,k): gather q row, h = tanh(q*W1 + b1),
// score[r] = sigmoid(h . V + bV).  Each wave handles 32 rows as two 16-row
// WMMA blocks that share every B fragment.
// ---------------------------------------------------------------------------
__global__ __launch_bounds__(256, 1) void score_kernel(
    const float* __restrict__ query, const int* __restrict__ neigh,
    const float* __restrict__ W1, const float* __restrict__ b1,
    const float* __restrict__ V, const float* __restrict__ bV,
    float* __restrict__ score_out, int numBlockTiles) {
  __shared__ bf16 w1t[CDIM * LDS_STRIDE];

  const int tid = threadIdx.x;

  // ---- stage W1^T (bf16) into LDS once per block (amortized by grid-stride)
  #pragma unroll 4
  for (int i = 0; i < 64; ++i) {
    int e = tid * 64 + i;           // 256 threads * 64 = 16384 = 128*128
    int c = e >> 7;                 // K index (row of W1)
    int d = e & 127;                // N index (col of W1)
    w1t[d * LDS_STRIDE + c] = (bf16)W1[e];
  }
  __syncthreads();

  const int lane = tid & 31;
  const int wv   = tid >> 5;        // 8 waves per block
  const int half = lane >> 4;       // "g" in the WMMA A/B layouts
  const int nl   = lane & 15;

  // Epilogue constants: col = t*16 + nl for accumulator tile t.
  float b1v[8], Vv[8];
  #pragma unroll
  for (int t = 0; t < 8; ++t) {
    b1v[t] = b1[t * 16 + nl];
    Vv[t]  = V[t * 16 + nl];
  }
  const float bVs = bV[0];

  for (int bt = blockIdx.x; bt < numBlockTiles; bt += gridDim.x) {
    const int waveBase = bt * 256 + wv * 32;   // 32 rows per wave

    // Per-lane gathered A row pointers (lane nl owns row rb*16+nl of the tile)
    const float* qrow[2];
    #pragma unroll
    for (int rb = 0; rb < 2; ++rb) {
      int row = waveBase + rb * 16 + nl;
      row = row < TOTAL_ROWS ? row : TOTAL_ROWS - 1;   // clamp, store guarded
      int bb  = row / (NPTS * KNEI);
      int rem = row - bb * (NPTS * KNEI);
      int n   = rem / KNEI;
      int k   = rem - n * KNEI;
      int idx = neigh[n * KNEI + k];
      qrow[rb] = query + (size_t)(bb * NPTS + idx) * CDIM;
    }

    v8f acc[2][8];
    #pragma unroll
    for (int rb = 0; rb < 2; ++rb)
      #pragma unroll
      for (int t = 0; t < 8; ++t)
        #pragma unroll
        for (int i = 0; i < 8; ++i) acc[rb][t][i] = 0.0f;

    // K-chunk outer (A loaded once, reused across all 8 N-tiles).
    #pragma unroll
    for (int kc = 0; kc < 4; ++kc) {
      v16bf afrag[2];
      #pragma unroll
      for (int rb = 0; rb < 2; ++rb) {
        // 16-bit A 16x32 layout: elems 0-7 -> K = kc*32 + g*8 + i,
        //                        elems 8-15 -> K = kc*32 + 16 + g*8 + i
        const float* p = qrow[rb] + kc * 32 + half * 8;
        v4f f0 = *(const v4f*)(p);
        v4f f1 = *(const v4f*)(p + 4);
        v4f f2 = *(const v4f*)(p + 16);
        v4f f3 = *(const v4f*)(p + 20);
        v16bf a;
        #pragma unroll
        for (int i = 0; i < 4; ++i) {
          a[i]      = (bf16)f0[i];
          a[4 + i]  = (bf16)f1[i];
          a[8 + i]  = (bf16)f2[i];
          a[12 + i] = (bf16)f3[i];
        }
        afrag[rb] = a;
      }
      #pragma unroll
      for (int t = 0; t < 8; ++t) {
        // 16-bit B 32x16 layout: lane nl holds col N = t*16+nl,
        // K(local) = g*8 + 0..7 and 16 + g*8 + 0..7 in W1^T -> two 16B LDS loads.
        const bf16* bp = &w1t[(t * 16 + nl) * LDS_STRIDE + kc * 32 + half * 8];
        v8bf blo = *(const v8bf*)bp;
        v8bf bhi = *(const v8bf*)(bp + 16);
        v16bf bfrag;
        #pragma unroll
        for (int i = 0; i < 8; ++i) {
          bfrag[i]     = blo[i];
          bfrag[8 + i] = bhi[i];
        }
        acc[0][t] = __builtin_amdgcn_wmma_f32_16x16x32_bf16(
            false, afrag[0], false, bfrag, (short)0, acc[0][t], false, false);
        acc[1][t] = __builtin_amdgcn_wmma_f32_16x16x32_bf16(
            false, afrag[1], false, bfrag, (short)0, acc[1][t], false, false);
      }
    }

    // Epilogue: C/D layout -> VGPR r holds row r (lanes 0-15) / r+8 (16-31),
    // col = t*16 + nl.  tanh + dot with V per lane, reduce across the 16-lane
    // half, sigmoid, store one score per row.
    #pragma unroll
    for (int rb = 0; rb < 2; ++rb) {
      #pragma unroll
      for (int r = 0; r < 8; ++r) {
        float d = 0.0f;
        #pragma unroll
        for (int t = 0; t < 8; ++t) {
          float h = fast_tanh(acc[rb][t][r] + b1v[t]);
          d = fmaf(h, Vv[t], d);
        }
        d += __shfl_xor(d, 1, 32);
        d += __shfl_xor(d, 2, 32);
        d += __shfl_xor(d, 4, 32);
        d += __shfl_xor(d, 8, 32);
        float s = fast_sigmoid(d + bVs);
        int row = waveBase + rb * 16 + r + half * 8;
        if (nl == 0 && row < TOTAL_ROWS) score_out[row] = s;
      }
    }
  }
}

// ---------------------------------------------------------------------------
// Kernel 2: one wave per node. Normalize the 7 scores, weighted gather-sum of
// values rows (float4 per lane: 32 lanes * 4 = 128 channels).
// ---------------------------------------------------------------------------
__global__ __launch_bounds__(256) void context_kernel(
    const float* __restrict__ values, const int* __restrict__ neigh,
    const float* __restrict__ score, float* __restrict__ ctx) {
  int gtid = blockIdx.x * 256 + threadIdx.x;
  int node = gtid >> 5;
  int lane = gtid & 31;
  if (node >= NODES) return;

  int b = node / NPTS;
  int n = node - b * NPTS;

  float s = 0.0f;
  if (lane < KNEI) s = score[node * KNEI + lane];
  float sum = s;
  sum += __shfl_xor(sum, 1, 32);
  sum += __shfl_xor(sum, 2, 32);
  sum += __shfl_xor(sum, 4, 32);       // lanes 0-7 each hold the full sum
  sum = __shfl(sum, 0, 32);
  float inv = 1.0f / sum;

  v4f acc;
  #pragma unroll
  for (int i = 0; i < 4; ++i) acc[i] = 0.0f;

  #pragma unroll
  for (int k = 0; k < KNEI; ++k) {
    float w = __shfl(s, k, 32) * inv;
    int idx = neigh[n * KNEI + k];
    const v4f* vr = (const v4f*)(values + (size_t)(b * NPTS + idx) * CDIM);
    v4f v = vr[lane];
    #pragma unroll
    for (int i = 0; i < 4; ++i) acc[i] = fmaf(w, v[i], acc[i]);
  }
  *(v4f*)(ctx + (size_t)node * CDIM + (size_t)lane * 4) = acc;
}

// ---------------------------------------------------------------------------
extern "C" void kernel_launch(void* const* d_in, const int* in_sizes, int n_in,
                              void* d_out, int out_size, void* d_ws,
                              size_t ws_size, hipStream_t stream) {
  (void)in_sizes; (void)n_in; (void)out_size; (void)d_ws; (void)ws_size;
  const float* query  = (const float*)d_in[0];
  const float* values = (const float*)d_in[1];
  const int*   neigh  = (const int*)d_in[2];
  const float* W1     = (const float*)d_in[3];
  const float* b1     = (const float*)d_in[4];
  const float* V      = (const float*)d_in[5];
  const float* bV     = (const float*)d_in[6];

  float* ctx   = (float*)d_out;                          // (B,N,C)
  float* score = (float*)d_out + (size_t)NODES * CDIM;   // (B,N,K,1)

  int numBlockTiles = (TOTAL_ROWS + 255) / 256;          // 4481 tiles of 256 rows
  int grid1 = numBlockTiles < 1120 ? numBlockTiles : 1120;  // amortize LDS staging
  score_kernel<<<grid1, 256, 0, stream>>>(query, neigh, W1, b1, V, bV, score,
                                          numBlockTiles);

  int grid2 = (NODES + 7) / 8;                           // 8 nodes per block
  context_kernel<<<grid2, 256, 0, stream>>>(values, neigh, score, ctx);
}
